// GIN_72584947302883
// MI455X (gfx1250) — matrix operand order repaired and checked by reading, not compiled, not run on hardware
//
#include <hip/hip_runtime.h>
#include <hip/hip_bf16.h>

// ---------------------------------------------------------------------------
// GIN 3-layer forward on MI455X (gfx1250, wave32).
//   agg = h + segment_sum(h[src], dst)       -> fp32 atomics (memory-bound)
//   y   = agg @ w1 + b1                      -> v_wmma_f32_16x16x32_f16
//   z   = relu(BN(y))                        -> 2-pass column stats + fused norm
//   h'  = z @ w2 + b2 [ -> relu(BN(.)) ]     -> v_wmma_f32_16x16x32_f16
// D = 256, N = 100000 (N/16 = 6250 exact -> no ragged WMMA tiles).
// ---------------------------------------------------------------------------

typedef __attribute__((ext_vector_type(16))) _Float16 v16h;
typedef __attribute__((ext_vector_type(8)))  _Float16 v8h;
typedef __attribute__((ext_vector_type(8)))  float    v8f;
typedef __attribute__((ext_vector_type(4)))  float    v4f;

#define DD 256
#define BN_EPS 1e-5f

// ---------------- copy h -> agg (float4) -----------------------------------
__global__ __launch_bounds__(256) void gin_copy_f4(const v4f* __restrict__ in,
                                                   v4f* __restrict__ out, long n4) {
  long i = (long)blockIdx.x * blockDim.x + threadIdx.x;
  if (i < n4) out[i] = in[i];
}

// ---------------- edge scatter: agg[dst] += h[src] --------------------------
// One block = 256 threads = one full feature row per edge; EPB edges per block.
__global__ __launch_bounds__(256) void gin_scatter(const float* __restrict__ h,
                                                   const int* __restrict__ src,
                                                   const int* __restrict__ dst,
                                                   float* __restrict__ agg,
                                                   int nEdges, int epb) {
  int t  = threadIdx.x;           // feature index 0..255
  int e0 = blockIdx.x * epb;
  for (int i = 0; i < epb; ++i) {
    int e = e0 + i;
    if (e >= nEdges) return;                    // uniform across block
    int s = src[e];
    int d = dst[e];
    atomicAdd(&agg[(long)d * DD + t], h[(long)s * DD + t]);
  }
}

// ---------------- weight transpose + f32->f16 -------------------------------
// Wt[col][k] = (f16) W[k][col]  -> B fragments become contiguous loads.
__global__ __launch_bounds__(256) void gin_wt_conv(const float* __restrict__ W,
                                                   _Float16* __restrict__ Wt) {
  int idx = blockIdx.x * 256 + threadIdx.x;     // idx = k*256 + col
  int k = idx >> 8;
  int c = idx & 255;
  Wt[(long)c * DD + k] = (_Float16)W[idx];
}

// ---------------- WMMA GEMM: C = A(Mx256) * W(256x256) + bias ---------------
// Wave tile: 16 rows x 64 cols (4x v_wmma_f32_16x16x32_f16, K-loop 8 steps).
// A fragment per ISA 16-bit A 16x32 layout, B per 32x16 striping, C per
// 32-bit C/D 16x16 layout. EXEC is all-ones at every WMMA (wave-level guard).
__global__ __launch_bounds__(256) void gin_gemm_bias(const float* __restrict__ A,
                                                     const _Float16* __restrict__ Wt,
                                                     const float* __restrict__ bias,
                                                     float* __restrict__ C,
                                                     int mTiles) {
  const int lane   = threadIdx.x & 31;
  const int waveId = threadIdx.x >> 5;
  const int gtile  = blockIdx.x * 8 + waveId;     // 8 waves / block
  const int total  = mTiles * 4;                  // 4 column groups of 64
  if (gtile >= total) return;                     // wave-uniform

  const int tm  = gtile >> 2;                     // M tile (16 rows)
  const int tn4 = gtile & 3;                      // 64-col group

  const int lrow = lane & 15;
  const int hi   = lane >> 4;                     // 0: lanes 0-15, 1: lanes 16-31
  const long row = (long)tm * 16 + lrow;
  const float* arow = A + row * DD;

  v8f acc0 = {}, acc1 = {}, acc2 = {}, acc3 = {};

#pragma unroll
  for (int k0 = 0; k0 < DD; k0 += 32) {
    // ---- A fragment: lanes 0-15 -> K {0..7,16..23}; lanes 16-31 -> {8..15,24..31}
    const int kb = k0 + (hi << 3);
    const v4f* ap0 = (const v4f*)(arow + kb);
    const v4f* ap1 = (const v4f*)(arow + kb + 16);
    v4f a0 = ap0[0], a1 = ap0[1];
    v4f a2 = ap1[0], a3 = ap1[1];
    v16h af;
#pragma unroll
    for (int i = 0; i < 4; ++i) {
      af[i]      = (_Float16)a0[i];
      af[4 + i]  = (_Float16)a1[i];
      af[8 + i]  = (_Float16)a2[i];
      af[12 + i] = (_Float16)a3[i];
    }

    // ---- B fragments: lanes 0-15 -> K {0..7,16..23}; lanes 16-31 -> {8..15,24..31}
    const int kcol = k0 + (hi << 3);
#pragma unroll
    for (int j = 0; j < 4; ++j) {
      const int col = tn4 * 64 + j * 16 + lrow;
      const v8h* bp = (const v8h*)(Wt + (long)col * DD + kcol);
      v8h b0 = bp[0], b1 = bp[2];
      v16h bf;
#pragma unroll
      for (int i = 0; i < 8; ++i) { bf[i] = b0[i]; bf[8 + i] = b1[i]; }

      v8f* accp = (j == 0) ? &acc0 : (j == 1) ? &acc1 : (j == 2) ? &acc2 : &acc3;
      *accp = __builtin_amdgcn_wmma_f32_16x16x32_f16(
          /*neg_a=*/false, af, /*neg_b=*/false, bf,
          /*c_mod=*/(short)0, *accp, /*reuse_a=*/false, /*reuse_b=*/false);
    }
  }

  // ---- epilogue: +bias, store. Lane L: vgpr i -> row (hi*8+i), col L&15.
  const int rOff = hi * 8;
#pragma unroll
  for (int j = 0; j < 4; ++j) {
    const int col = tn4 * 64 + j * 16 + lrow;
    const float bv = bias[col];
    const v8f acc = (j == 0) ? acc0 : (j == 1) ? acc1 : (j == 2) ? acc2 : acc3;
#pragma unroll
    for (int i = 0; i < 8; ++i) {
      long r = (long)tm * 16 + rOff + i;
      C[r * DD + col] = acc[i] + bv;
    }
  }
}

// ---------------- zero the 512-float stats buffer ---------------------------
__global__ __launch_bounds__(512) void gin_zero_stats(float* __restrict__ stats) {
  stats[threadIdx.x] = 0.0f;
}

// ---------------- per-column sum / sumsq over rows --------------------------
__global__ __launch_bounds__(256) void gin_colstats(const float* __restrict__ Y,
                                                    float* __restrict__ stats,
                                                    int mRows, int rowsPerBlock) {
  const int c  = threadIdx.x;                 // column 0..255
  const int r0 = blockIdx.x * rowsPerBlock;
  int rend = r0 + rowsPerBlock;
  if (rend > mRows) rend = mRows;
  float s = 0.0f, s2 = 0.0f;
  for (int r = r0; r < rend; ++r) {
    float v = Y[(long)r * DD + c];
    s  += v;
    s2 += v * v;
  }
  atomicAdd(&stats[c], s);
  atomicAdd(&stats[DD + c], s2);
}

// ---------------- fused BN (training-mode, biased var) + ReLU ---------------
__global__ __launch_bounds__(256) void gin_bn_relu(const float* __restrict__ Y,
                                                   const float* __restrict__ stats,
                                                   const float* __restrict__ g,
                                                   const float* __restrict__ be,
                                                   float* __restrict__ out,
                                                   long total4, float invN) {
  long i4 = (long)blockIdx.x * blockDim.x + threadIdx.x;
  if (i4 >= total4) return;
  const int c0 = (int)((i4 * 4) & 255);
  v4f y = ((const v4f*)Y)[i4];
  v4f o;
#pragma unroll
  for (int j = 0; j < 4; ++j) {
    const int c = c0 + j;
    const float m  = stats[c] * invN;
    const float v  = stats[DD + c] * invN - m * m;
    const float rs = rsqrtf(v + BN_EPS);
    const float z  = (y[j] - m) * rs * g[c] + be[c];
    o[j] = z > 0.0f ? z : 0.0f;
  }
  ((v4f*)out)[i4] = o;
}

// ---------------------------------------------------------------------------
extern "C" void kernel_launch(void* const* d_in, const int* in_sizes, int n_in,
                              void* d_out, int out_size, void* d_ws, size_t ws_size,
                              hipStream_t stream) {
  const float* x   = (const float*)d_in[0];
  const int*   src = (const int*)d_in[1];
  const int*   dst = (const int*)d_in[2];

  const int  nNodes = in_sizes[0] / DD;       // 100000
  const int  nEdges = in_sizes[1];            // 1600000
  const long ND     = (long)nNodes * DD;
  const float invN  = 1.0f / (float)nNodes;

  // workspace layout
  float* bufP = (float*)d_ws;                 // h ping (layer outputs 0,1)
  float* bufR = bufP + ND;                    // agg / normalized z
  float* bufS = bufR + ND;                    // y / h2 scratch
  float* stats = bufS + ND;                   // 512 floats
  _Float16* wt1 = (_Float16*)(stats + 512);   // 256x256 f16 (transposed)
  _Float16* wt2 = wt1 + DD * DD;

  const int mTiles    = nNodes / 16;          // 6250
  const int gemmGrid  = (mTiles * 4 + 7) / 8; // 8 wave-tiles per block
  const long n4       = ND / 4;
  const int copyGrid  = (int)((n4 + 255) / 256);
  const int epb       = 8;
  const int scatGrid  = (nEdges + epb - 1) / epb;
  const int statsRows = 128;
  const int statGrid  = (nNodes + statsRows - 1) / statsRows;
  const long t4       = ND / 4;
  const int bnGrid    = (int)((t4 + 255) / 256);

  const float* hin = x;
  for (int l = 0; l < 3; ++l) {
    const int base = 3 + l * 8;
    const float* w1  = (const float*)d_in[base + 0];
    const float* b1  = (const float*)d_in[base + 1];
    const float* g1  = (const float*)d_in[base + 2];
    const float* be1 = (const float*)d_in[base + 3];
    const float* w2  = (const float*)d_in[base + 4];
    const float* b2  = (const float*)d_in[base + 5];

    // 1) agg = h ; agg[dst] += h[src]
    gin_copy_f4<<<copyGrid, 256, 0, stream>>>((const v4f*)hin, (v4f*)bufR, n4);
    gin_scatter<<<scatGrid, 256, 0, stream>>>(hin, src, dst, bufR, nEdges, epb);

    // 2) transpose+convert weights to f16 [col][k]
    gin_wt_conv<<<DD, 256, 0, stream>>>(w1, wt1);
    gin_wt_conv<<<DD, 256, 0, stream>>>(w2, wt2);

    // 3) y = agg @ w1 + b1   (WMMA)
    gin_gemm_bias<<<gemmGrid, 256, 0, stream>>>(bufR, wt1, b1, bufS, mTiles);

    // 4) z = relu(BN(y)) -> bufR
    gin_zero_stats<<<1, 512, 0, stream>>>(stats);
    gin_colstats<<<statGrid, 256, 0, stream>>>(bufS, stats, nNodes, statsRows);
    gin_bn_relu<<<bnGrid, 256, 0, stream>>>(bufS, stats, g1, be1, bufR, t4, invN);

    // 5) h2 = z @ w2 + b2   (WMMA); layer 2 writes straight to d_out
    float* out2 = (l < 2) ? bufS : (float*)d_out;
    gin_gemm_bias<<<gemmGrid, 256, 0, stream>>>(bufR, wt2, b2, out2, mTiles);

    // 6) layers 0,1: h' = relu(BN(h2)) -> bufP
    if (l < 2) {
      const float* g2  = (const float*)d_in[base + 6];
      const float* be2 = (const float*)d_in[base + 7];
      gin_zero_stats<<<1, 512, 0, stream>>>(stats);
      gin_colstats<<<statGrid, 256, 0, stream>>>(bufS, stats, nNodes, statsRows);
      gin_bn_relu<<<bnGrid, 256, 0, stream>>>(bufS, stats, g2, be2, bufP, t4, invN);
      hin = bufP;
    }
  }
}
